// analytic_entropy_45810121179237
// MI455X (gfx1250) — compile-verified, matrix operand order checked by reading, not run on hardware
//
#include <hip/hip_runtime.h>
#include <hip/hip_bf16.h>

// ---------------- types ----------------
typedef __attribute__((ext_vector_type(16))) __bf16 v16bf;
typedef __attribute__((ext_vector_type(2)))  __bf16 v2bf;
typedef __attribute__((ext_vector_type(8)))  float  v8f;

#define NROWS 8192
#define KDIM  1024
#define SIGMA 0.3f

#define BM 128      // block tile rows (ex)
#define BN 256      // block tile cols (ey)
#define BK 32       // k-chunk
#define ROWB 80     // LDS bytes per 32-k row: 64B data + 16B pad (conflict-free b128)

// order-preserving float<->uint mapping (for atomic max over signed floats)
__device__ __forceinline__ unsigned fordered(float f) {
    unsigned u = __float_as_uint(f);
    return (u & 0x80000000u) ? ~u : (u | 0x80000000u);
}
__device__ __forceinline__ float funordered(unsigned u) {
    unsigned v = (u & 0x80000000u) ? (u & 0x7FFFFFFFu) : ~u;
    return __uint_as_float(v);
}

// f32 pair -> packed bf16.
#if __has_builtin(__builtin_amdgcn_cvt_pk_bf16_f32)
__device__ __forceinline__ unsigned pk_bf16(float a, float b) {
    v2bf p = __builtin_amdgcn_cvt_pk_bf16_f32(a, b);
    unsigned r;
    __builtin_memcpy(&r, &p, 4);
    return r;
}
#else
// round-half-up then splice the two high halves with one v_perm_b32:
// dst = { b[3], b[2], a[3], a[2] }  (3 VALU total)
__device__ __forceinline__ unsigned pk_bf16(float a, float b) {
    unsigned ua = __float_as_uint(a) + 0x8000u;
    unsigned ub = __float_as_uint(b) + 0x8000u;
    return __builtin_amdgcn_perm(ub, ua, 0x07060302u);
}
#endif

__device__ __forceinline__ uint4 pk4(float4 a, float4 b) {
    uint4 r;
    r.x = pk_bf16(a.x, a.y); r.y = pk_bf16(a.z, a.w);
    r.z = pk_bf16(b.x, b.y); r.w = pk_bf16(b.z, b.w);
    return r;
}

union FragU { uint4 q[2]; v16bf v; };

// ---------------- kernels ----------------

// zero the 2*8192 ordered-uint max arrays (identity for max = 0 in ordered space)
__global__ void ce_init_kernel(unsigned* p) {
    p[blockIdx.x * 256 + threadIdx.x] = 0u;
}

// per-row L2 norms of ex (y=0) and ey (y=1); one wave per row
__global__ __launch_bounds__(256) void ce_norm_kernel(const float* __restrict__ ex,
                                                      const float* __restrict__ ey,
                                                      float* __restrict__ nx,
                                                      float* __restrict__ ny) {
    const float* src = blockIdx.y ? ey : ex;
    float*       dst = blockIdx.y ? ny : nx;
    int wave = threadIdx.x >> 5, lane = threadIdx.x & 31;
    int row  = blockIdx.x * 8 + wave;
    const float* p = src + (size_t)row * KDIM;
    float s = 0.f;
#pragma unroll
    for (int c = 0; c < 8; ++c) {
        float4 v = *(const float4*)(p + lane * 4 + c * 128);
        s += v.x * v.x + v.y * v.y + v.z * v.z + v.w * v.w;
    }
#pragma unroll
    for (int m = 16; m; m >>= 1) s += __shfl_xor(s, m);
    if (lane == 0) dst[row] = sqrtf(s);
}

// WMMA GEMM tile + cosine scaling + row/col max reduction
__global__ __launch_bounds__(256) void ce_gemm_max_kernel(const float* __restrict__ ex,
                                                          const float* __restrict__ ey,
                                                          const float* __restrict__ nx,
                                                          const float* __restrict__ ny,
                                                          unsigned* __restrict__ rowMax,
                                                          unsigned* __restrict__ colMax) {
    __shared__ __align__(16) char As[BM * ROWB];   // 10240 B  bf16 A plane
    __shared__ __align__(16) char Bs[BN * ROWB];   // 20480 B  bf16 B plane
    __shared__ unsigned redRow[BM];
    __shared__ unsigned redCol[BN];

    const int tid  = threadIdx.x;
    const int wave = tid >> 5;
    const int lane = tid & 31;
    const int wr   = wave & 1;    // 0..1 -> 64-row strip
    const int wc   = wave >> 1;   // 0..3 -> 64-col strip
    const int lg   = lane >> 4;   // lane group (k/M offset select)
    const int ln   = lane & 15;

    const int rowBase = blockIdx.y * BM;
    const int colBase = blockIdx.x * BN;

    for (int i = tid; i < BM; i += 256) redRow[i] = 0u;
    for (int i = tid; i < BN; i += 256) redCol[i] = 0u;

    v8f acc[4][4];
#pragma unroll
    for (int i = 0; i < 4; ++i)
#pragma unroll
        for (int j = 0; j < 4; ++j) acc[i][j] = (v8f){};

    // cooperative loader mapping
    const int arow = tid >> 1, ahalf = tid & 1;          // A: 128 rows x (2 halves of 16 k)
    const float* aptr = ex + (size_t)(rowBase + arow) * KDIM + ahalf * 16;
    const float* bptr = ey + (size_t)(colBase + tid) * KDIM;   // B: 256 rows x 32 k

    for (int k0 = 0; k0 < KDIM; k0 += BK) {
        float4 av[4], bv[8];
#pragma unroll
        for (int c = 0; c < 4; ++c) av[c] = *(const float4*)(aptr + k0 + c * 4);
#pragma unroll
        for (int c = 0; c < 8; ++c) bv[c] = *(const float4*)(bptr + k0 + c * 4);

        __syncthreads();   // previous chunk's LDS reads complete (WAR)

        uint4* adst = (uint4*)(As + arow * ROWB + ahalf * 32);
        adst[0] = pk4(av[0], av[1]);
        adst[1] = pk4(av[2], av[3]);
        uint4* bdst = (uint4*)(Bs + tid * ROWB);
#pragma unroll
        for (int c = 0; c < 4; ++c) bdst[c] = pk4(bv[2 * c], bv[2 * c + 1]);

        __syncthreads();   // LDS stores visible (RAW)

        // per-lane fragment fetch: 16 bf16 = two b128 at +0 and +32 bytes
        v16bf a[4], b[4];
#pragma unroll
        for (int i = 0; i < 4; ++i) {
            const uint4* p = (const uint4*)(As + (wr * 64 + i * 16 + ln) * ROWB + lg * 16);
            FragU u; u.q[0] = p[0]; u.q[1] = p[2];
            a[i] = u.v;
        }
#pragma unroll
        for (int j = 0; j < 4; ++j) {
            const uint4* p = (const uint4*)(Bs + (wc * 64 + j * 16 + ln) * ROWB + lg * 16);
            FragU u; u.q[0] = p[0]; u.q[1] = p[2];
            b[j] = u.v;
        }
#pragma unroll
        for (int i = 0; i < 4; ++i)
#pragma unroll
            for (int j = 0; j < 4; ++j)
                acc[i][j] = __builtin_amdgcn_wmma_f32_16x16x32_bf16(
                    false, a[i], false, b[j], (short)0, acc[i][j], false, false);
    }

    // ---- epilogue: cosine scaling + maxima ----
    float nxv[4][8];
#pragma unroll
    for (int i = 0; i < 4; ++i) {
        const float4* p = (const float4*)(nx + rowBase + wr * 64 + i * 16 + lg * 8);
        float4 x0 = p[0], x1 = p[1];
        nxv[i][0] = x0.x; nxv[i][1] = x0.y; nxv[i][2] = x0.z; nxv[i][3] = x0.w;
        nxv[i][4] = x1.x; nxv[i][5] = x1.y; nxv[i][6] = x1.z; nxv[i][7] = x1.w;
    }
    float nyv[4];
#pragma unroll
    for (int j = 0; j < 4; ++j) nyv[j] = ny[colBase + wc * 64 + j * 16 + ln];

    float rmax[4][8];
    float cmax[4];
#pragma unroll
    for (int i = 0; i < 4; ++i)
#pragma unroll
        for (int e = 0; e < 8; ++e) rmax[i][e] = -3.0e38f;
#pragma unroll
    for (int j = 0; j < 4; ++j) cmax[j] = -3.0e38f;

#pragma unroll
    for (int i = 0; i < 4; ++i)
#pragma unroll
        for (int j = 0; j < 4; ++j)
#pragma unroll
            for (int e = 0; e < 8; ++e) {
                float d = fmaxf(nxv[i][e] * nyv[j], 1e-8f);
                float v = acc[i][j][e] * __builtin_amdgcn_rcpf(d);
                rmax[i][e] = fmaxf(rmax[i][e], v);
                cmax[j]    = fmaxf(cmax[j], v);
            }

    // row maxima: reduce across the 16 lanes of a half-wave (same M, N=0..15)
#pragma unroll
    for (int i = 0; i < 4; ++i)
#pragma unroll
        for (int e = 0; e < 8; ++e) {
            float v = rmax[i][e];
            v = fmaxf(v, __shfl_xor(v, 1));
            v = fmaxf(v, __shfl_xor(v, 2));
            v = fmaxf(v, __shfl_xor(v, 4));
            v = fmaxf(v, __shfl_xor(v, 8));
            if (ln == 0)
                atomicMax(&redRow[wr * 64 + i * 16 + lg * 8 + e], fordered(v));
        }
    // col maxima: combine the two half-wave M-groups, then per-lane N
#pragma unroll
    for (int j = 0; j < 4; ++j) {
        float v = cmax[j];
        v = fmaxf(v, __shfl_xor(v, 16));
        if (lg == 0)
            atomicMax(&redCol[wc * 64 + j * 16 + ln], fordered(v));
    }

    __syncthreads();
    for (int i = tid; i < BM; i += 256) atomicMax(&rowMax[rowBase + i], redRow[i]);
    for (int i = tid; i < BN; i += 256) atomicMax(&colMax[colBase + i], redCol[i]);
}

// apply H and sum both max arrays -> out[2]
__global__ __launch_bounds__(256) void ce_finish_kernel(const unsigned* __restrict__ rowMax,
                                                        const unsigned* __restrict__ colMax,
                                                        float* __restrict__ out) {
    __shared__ float red[256];
    const float HC = 0.5f * logf(6.283185307179586f * SIGMA * SIGMA) + 0.5f;
    const float scale = HC / SIGMA;

    float s1 = 0.f, s2 = 0.f;
    for (int i = threadIdx.x; i < NROWS; i += 256) {
        s1 += 1.0f - funordered(rowMax[i]);
        s2 += 1.0f - funordered(colMax[i]);
    }
    // reduce s1
    red[threadIdx.x] = s1; __syncthreads();
    for (int off = 128; off; off >>= 1) {
        if (threadIdx.x < off) red[threadIdx.x] += red[threadIdx.x + off];
        __syncthreads();
    }
    if (threadIdx.x == 0) out[0] = scale * red[0];
    __syncthreads();
    // reduce s2
    red[threadIdx.x] = s2; __syncthreads();
    for (int off = 128; off; off >>= 1) {
        if (threadIdx.x < off) red[threadIdx.x] += red[threadIdx.x + off];
        __syncthreads();
    }
    if (threadIdx.x == 0) out[1] = scale * red[0];
}

// ---------------- launch ----------------
extern "C" void kernel_launch(void* const* d_in, const int* in_sizes, int n_in,
                              void* d_out, int out_size, void* d_ws, size_t ws_size,
                              hipStream_t stream) {
    const float* ex = (const float*)d_in[0];
    const float* ey = (const float*)d_in[1];
    float* out = (float*)d_out;

    // workspace layout: nx[8192] f32 | ny[8192] f32 | rowMax[8192] u32 | colMax[8192] u32
    float*    nx     = (float*)d_ws;
    float*    ny     = nx + NROWS;
    unsigned* rowMax = (unsigned*)(ny + NROWS);
    unsigned* colMax = rowMax + NROWS;

    ce_init_kernel<<<(2 * NROWS) / 256, 256, 0, stream>>>(rowMax);
    ce_norm_kernel<<<dim3(NROWS / 8, 2), 256, 0, stream>>>(ex, ey, nx, ny);
    ce_gemm_max_kernel<<<dim3(NROWS / BN, NROWS / BM), 256, 0, stream>>>(
        ex, ey, nx, ny, rowMax, colMax);
    ce_finish_kernel<<<1, 256, 0, stream>>>(rowMax, colMax, out);
}
